// LSTMRecursiveModel_16045997818189
// MI455X (gfx1250) — matrix operand order repaired and checked by reading, not compiled, not run on hardware
//
#include <hip/hip_runtime.h>

typedef _Float16 h16;
typedef __attribute__((ext_vector_type(16))) _Float16 v16h;
typedef __attribute__((ext_vector_type(8)))  float    v8f;
typedef __attribute__((ext_vector_type(4)))  int      i32x4;

#define NWG 16
#define TPB 128

#define ASG __attribute__((address_space(1)))
#define ASL __attribute__((address_space(3)))

#if defined(__has_builtin)
#  if __has_builtin(__builtin_amdgcn_global_load_async_to_lds_b128) && \
      __has_builtin(__builtin_amdgcn_s_wait_asynccnt)
#    define USE_ASYNC_LDS 1
#  endif
#endif
#ifndef USE_ASYNC_LDS
#  define USE_ASYNC_LDS 0
#endif

constexpr int Bb = 16, Ll = 96, Hh = 256, Oo = 32;
constexpr int P0 = 272;    // layer-0 weight K pitch (halves), padded vs 256
constexpr int P1 = 528;    // layer-1 weight K pitch (halves), padded vs 512
constexpr int PAB = 528;   // activation buffer row pitch (halves)
constexpr int SEQP = 128;  // rolling sequence row pitch (floats): 96 x + 32 preds
constexpr int SMEM_BYTES =
    (4*16*P0 + 4*16*P1 + 16*PAB) * 2 +                       // f16 regions
    (4*256 + 16*SEQP + 2*256 + 2*64 + 64 + 256 + 4) * 4;     // f32 regions

__device__ __forceinline__ float sigf(float x) {
  return __builtin_amdgcn_rcpf(1.0f + __expf(-x));
}
__device__ __forceinline__ float tanh_(float x) {
  float e = __expf(2.0f * x);
  return 1.0f - 2.0f * __builtin_amdgcn_rcpf(e + 1.0f);
}

union Frag { v16h v; uint4 u[2]; };

// broadcast full h [16][256] f16 from global into AB[:, COLOFF .. COLOFF+256)
template <int COLOFF>
__device__ __forceinline__ void bcast(h16* AB, const h16* src, int tid) {
  #pragma unroll
  for (int u = tid; u < 512; u += TPB) {        // 512 x 16B units, 4 per thread
    int r = u >> 5, c8 = (u & 31) << 3;
    h16* d = AB + r*PAB + COLOFF + c8;
    const h16* s = src + r*Hh + c8;
#if USE_ASYNC_LDS
    __builtin_amdgcn_global_load_async_to_lds_b128(
        (ASG i32x4*)s, (ASL i32x4*)d, 0, 0);
#else
    *(uint4*)d = *(const uint4*)s;
#endif
  }
#if USE_ASYNC_LDS
  __builtin_amdgcn_s_wait_asynccnt(0);
#endif
}

__global__ __launch_bounds__(TPB, 1)
void lstm_persistent(const float* __restrict__ x_enc,
                     const float* __restrict__ Wih0, const float* __restrict__ Whh0,
                     const float* __restrict__ bih0, const float* __restrict__ bhh0,
                     const float* __restrict__ Wih1, const float* __restrict__ Whh1,
                     const float* __restrict__ bih1, const float* __restrict__ bhh1,
                     const float* __restrict__ fc_w, const float* __restrict__ fc_b,
                     float* __restrict__ out,
                     h16* __restrict__ gh0, h16* __restrict__ gh1,
                     unsigned* __restrict__ ctr0, unsigned* __restrict__ ctr1)
{
  extern __shared__ char smem[];
  h16*   Wt0  = (h16*)smem;                    // [4 gate][16 col][P0]   layer-0 Whh^T slice
  h16*   Wt1  = Wt0 + 4*16*P0;                 // [4][16][P1]            layer-1 [Whh|Wih]^T slice
  h16*   AB   = Wt1 + 4*16*P1;                 // [16 batch][PAB]: cols 0..255 h1, 256..511 h0
  float* gbuf = (float*)(AB + 16*PAB);         // [4][16][16] gate tiles
  float* seq  = gbuf + 4*256;                  // [16][SEQP] x window + preds
  float* cst  = seq + 16*SEQP;                 // [2][16][16] cell state (this WG's cols)
  float* bias = cst + 2*256;                   // [2][64] combined biases
  float* wx0  = bias + 2*64;                   // [64]  Wih0 column
  float* fcw  = wx0 + 64;                      // [256]
  float* fcbp = fcw + 256;                     // [1]

  const int tid = threadIdx.x;
  const int blk = blockIdx.x;                  // owns h columns [16*blk, 16*blk+16)
  const int wv  = tid >> 5;                    // gate group i/f/g/o
  const int ln  = tid & 31;

  // ---------------- one-time init (weights -> LDS, f32 -> f16) ----------------
  for (int i = tid; i < 4*16*256; i += TPB) {
    int g = i >> 12, j = (i >> 8) & 15, k = i & 255;
    int row = g*Hh + blk*16 + j;
    Wt0[(g*16 + j)*P0 + k] = (h16)Whh0[row*Hh + k];
  }
  for (int i = tid; i < 4*16*512; i += TPB) {
    int g = i >> 13, j = (i >> 9) & 15, k = i & 511;
    int row = g*Hh + blk*16 + j;
    float w = (k < 256) ? Whh1[row*Hh + k] : Wih1[row*Hh + (k - 256)];
    Wt1[(g*16 + j)*P1 + k] = (h16)w;
  }
  for (int i = tid; i < 64; i += TPB) {
    int row = (i >> 4)*Hh + blk*16 + (i & 15);
    bias[i]      = bih0[row] + bhh0[row];
    bias[64 + i] = bih1[row] + bhh1[row];
    wx0[i]       = Wih0[row];
  }
  for (int i = tid; i < 256; i += TPB) fcw[i] = fc_w[i];
  if (tid == 0) fcbp[0] = fc_b[0];
  for (int i = tid; i < 16*Ll; i += TPB) seq[(i / Ll)*SEQP + (i % Ll)] = x_enc[i];
  for (int i = tid; i < 16*PAB; i += TPB) AB[i] = (h16)0.0f;
  for (int i = tid; i < 2*256; i += TPB) cst[i] = 0.0f;
  if (blk == 0) for (int i = tid; i < 16*Ll; i += TPB) out[i] = 0.0f;
  __syncthreads();

  // WMMA lane mapping (16x16x32 f16, wave32)
  const int n     = ln & 15;                   // output column within tile
  const int mbase = (ln >> 4) << 3;            // C/D rows: 0..7 or 8..15
  const int kg    = (ln >> 4) << 3;            // B: K sub-offset 0/8
  const int ko    = (ln >> 4) << 3;            // A: K sub-offset 0/8
  const h16* w0base = Wt0 + (wv*16 + n)*P0 + kg;
  const h16* w1base = Wt1 + (wv*16 + n)*P1 + kg;
  const h16* arow   = AB + (ln & 15)*PAB + ko;

  auto spin = [&](unsigned* c, unsigned tgt) {
    if (tid == 0)
      while (__hip_atomic_load(c, __ATOMIC_ACQUIRE, __HIP_MEMORY_SCOPE_AGENT) < tgt)
        __builtin_amdgcn_s_sleep(2);
    __syncthreads();
  };
  auto arrive = [&](unsigned* c) {
    __syncthreads();
    if (tid == 0)
      __hip_atomic_fetch_add(c, 1u, __ATOMIC_RELEASE, __HIP_MEMORY_SCOPE_AGENT);
  };
  auto combine = [&](float* cs, h16* gh) {     // gates -> (c,h); h -> global f16
    for (int e = tid; e < 256; e += TPB) {
      float I = gbuf[e], F = gbuf[256 + e], G = gbuf[512 + e], Og = gbuf[768 + e];
      float c = sigf(F)*cs[e] + sigf(I)*tanh_(G);
      float h = sigf(Og)*tanh_(c);
      cs[e] = c;
      gh[(e >> 4)*Hh + blk*16 + (e & 15)] = (h16)h;
    }
  };
  auto emit_pred = [&](int s) {                // pred = h1 @ fc_w^T + fc_b
    if (tid < 16) {
      const h16* hr = AB + tid*PAB;
      float a = fcbp[0];
      for (int k = 0; k < 256; ++k) a += (float)hr[k] * fcw[k];
      seq[tid*SEQP + Ll + s] = a;
      if (blk == 0) out[tid*Ll + s] = a;
    }
    __syncthreads();
  };

  unsigned b0t = 0, b1t = 0;
  bool first = true;

  // 33 windows x 96 steps: w==0 warm-up over x_enc, w>=1 rolling window s=w-1.
  // Software-pipelined: the h1 barrier of step q is waited AFTER layer-0 of
  // step q+1, hiding one agent-scope barrier round-trip behind WMMA work.
  for (int w = 0; w <= 32; ++w) {
    const int xbase = (w == 0) ? 0 : (w - 1);
    for (int t = 0; t < 96; ++t) {
      // ---- layer 0: gates = x*Wih0 + h0_prev @ Whh0^T + b ----
      {
        float bn = bias[wv*16 + n], wn = wx0[wv*16 + n];
        v8f acc;
        #pragma unroll
        for (int v = 0; v < 8; ++v)
          acc[v] = bn + seq[(mbase + v)*SEQP + (xbase + t)] * wn;
        #pragma unroll
        for (int k = 0; k < 8; ++k) {
          Frag fa, fb;
          fa.u[0] = *(const uint4*)(arow + 256 + 32*k);
          fa.u[1] = *(const uint4*)(arow + 256 + 32*k + 16);
          fb.u[0] = *(const uint4*)(w0base + 32*k);
          fb.u[1] = *(const uint4*)(w0base + 32*k + 16);
          acc = __builtin_amdgcn_wmma_f32_16x16x32_f16(false, fa.v, false, fb.v,
                                                       (short)0, acc, false, false);
        }
        #pragma unroll
        for (int v = 0; v < 8; ++v) gbuf[wv*256 + (mbase + v)*16 + n] = acc[v];
      }
      __syncthreads();
      combine(cst, gh0);                       // h0(t) -> global
      arrive(ctr0); b0t += NWG;

      // ---- deferred: finish previous step's h1 (overlapped with layer 0) ----
      if (!first) {
        spin(ctr1, b1t);
        bcast<0>(AB, gh1, tid);                // h1(t-1) -> AB cols 0..255
        if (t == 0 && w >= 2) {
          __syncthreads();
          emit_pred(w - 2);                    // window (w-2) ended at previous step
        }
      }
      first = false;

      // ---- h0 broadcast ----
      spin(ctr0, b0t);
      bcast<256>(AB, gh0, tid);                // h0(t) -> AB cols 256..511
      __syncthreads();

      // ---- layer 1: gates = [h1_prev | h0_cur] @ [Whh1 | Wih1]^T + b ----
      {
        float bn = bias[64 + wv*16 + n];
        v8f acc;
        #pragma unroll
        for (int v = 0; v < 8; ++v) acc[v] = bn;
        #pragma unroll
        for (int k = 0; k < 16; ++k) {
          Frag fa, fb;
          fa.u[0] = *(const uint4*)(arow + 32*k);
          fa.u[1] = *(const uint4*)(arow + 32*k + 16);
          fb.u[0] = *(const uint4*)(w1base + 32*k);
          fb.u[1] = *(const uint4*)(w1base + 32*k + 16);
          acc = __builtin_amdgcn_wmma_f32_16x16x32_f16(false, fa.v, false, fb.v,
                                                       (short)0, acc, false, false);
        }
        #pragma unroll
        for (int v = 0; v < 8; ++v) gbuf[wv*256 + (mbase + v)*16 + n] = acc[v];
      }
      __syncthreads();
      combine(cst + 256, gh1);                 // h1(t) -> global
      arrive(ctr1); b1t += NWG;
    }
  }

  // drain: final h1 barrier + last prediction (s = 31)
  spin(ctr1, b1t);
  bcast<0>(AB, gh1, tid);
  __syncthreads();
  emit_pred(31);
}

extern "C" void kernel_launch(void* const* d_in, const int* in_sizes, int n_in,
                              void* d_out, int out_size, void* d_ws, size_t ws_size,
                              hipStream_t stream) {
  (void)in_sizes; (void)n_in; (void)out_size; (void)ws_size;
  const float* x_enc = (const float*)d_in[0];
  const float* Wih0  = (const float*)d_in[4];
  const float* Whh0  = (const float*)d_in[5];
  const float* bih0  = (const float*)d_in[6];
  const float* bhh0  = (const float*)d_in[7];
  const float* Wih1  = (const float*)d_in[8];
  const float* Whh1  = (const float*)d_in[9];
  const float* bih1  = (const float*)d_in[10];
  const float* bhh1  = (const float*)d_in[11];
  const float* fc_w  = (const float*)d_in[12];
  const float* fc_b  = (const float*)d_in[13];

  h16* gh0 = (h16*)d_ws;                          // [16][256] f16
  h16* gh1 = gh0 + 16*256;                        // [16][256] f16
  unsigned* ctr0 = (unsigned*)((char*)d_ws + 16384);
  unsigned* ctr1 = ctr0 + 16;
  (void)hipMemsetAsync((char*)d_ws + 16384, 0, 256, stream); // reset barriers (capture-safe)

  hipLaunchKernelGGL(lstm_persistent, dim3(NWG), dim3(TPB), SMEM_BYTES, stream,
                     x_enc, Wih0, Whh0, bih0, bhh0, Wih1, Whh1, bih1, bhh1,
                     fc_w, fc_b, (float*)d_out, gh0, gh1, ctr0, ctr1);
}
